// MessagePassing_9740985827683
// MI455X (gfx1250) — matrix operand order repaired and checked by reading, not compiled, not run on hardware
//
#include <hip/hip_runtime.h>

// MessagePassing GNN layer for MI455X (gfx1250), wave32 + WMMA.
//
// out[n] = sum_{edges e with seg[e]==n} (nodes@W_node)[index[e]] *
//          LReLU(LReLU(edges@W_e1+b1)@W_e2+b2)
//
// f16 A/B, f32 accumulate via v_wmma_f32_16x16x32_f16.
// Weights pre-swizzled in LDS to fragment-major layout (B-fragment = one
// contiguous 32B -> 2x ds_load_b128), preloaded per n-tile into register
// arrays so dscnt waits stage instead of serializing each WMMA. msg/out fit
// in 192MB L2 so the fused gather+scatter atomics resolve on-chip. All
// element offsets fit int32 -> SADDR + 32-bit VGPR scale_offset addressing.
// Tile-edge handling is branch-free index clamping (N, E are multiples of 16
// here, so clamps never fire; they only guarantee no OOB in general).

#define NODE_CH 128
#define EDGE_CH 64
#define HID     128
#define NEGSLOPE 0.01f
#define EWAVES 8
#define NWAVES 8

typedef __attribute__((ext_vector_type(16))) _Float16 v16h;
typedef __attribute__((ext_vector_type(8)))  _Float16 v8h;
typedef __attribute__((ext_vector_type(8)))  float    v8f;

__device__ __forceinline__ v8f wmma_f16(v16h a, v16h b, v8f c) {
  // (neg_a, A, neg_b, B, c_mod, C, reuse_a, reuse_b)
  return __builtin_amdgcn_wmma_f32_16x16x32_f16(false, a, false, b, (short)0, c,
                                                false, false);
}

__device__ __forceinline__ float lrelu(float x) {
  return (x >= 0.f) ? x : NEGSLOPE * x;
}

__device__ __forceinline__ int imin(int a, int b) { return a < b ? a : b; }

// ---- A fragment (16-bit A 16x32, ISA 7.12.2) ----
// Lane (hi, r): halfs = row[k0 + 8*hi + 0..7] ++ row[k0 + 16 + 8*hi + 0..7]
__device__ __forceinline__ v16h load_a_f32v(const float* __restrict__ row,
                                            int k0, int hi) {
  v8f lo = *(const v8f*)(row + k0 + (hi << 3));
  v8f hh = *(const v8f*)(row + k0 + 16 + (hi << 3));
  v16h a;
#pragma unroll
  for (int i = 0; i < 8; ++i) {
    a[i]     = (_Float16)lo[i];
    a[8 + i] = (_Float16)hh[i];
  }
  return a;
}

__device__ __forceinline__ v16h load_a_f16v(const _Float16* __restrict__ row,
                                            int k0, int hi) {
  v8h lo = *(const v8h*)(row + k0 + (hi << 3));
  v8h hh = *(const v8h*)(row + k0 + 16 + (hi << 3));
  return __builtin_shufflevector(lo, hh, 0, 1, 2, 3, 4, 5, 6, 7, 8, 9, 10, 11,
                                 12, 13, 14, 15);
}

// ---- B fragments, fragment-major in LDS ----
// dst[((n*NKT + kt)*32 + lane)*16 + pos] = (f16)W[kt*32 + 8*(lane>>4) + (pos&7) + 16*(pos>>3)][n*16 + (lane&15)]
template <int NKT>
__device__ __forceinline__ void fill_b_frags(_Float16* dst,
                                             const float* __restrict__ W,
                                             int tid, int nthreads) {
  const int total = (NKT * 32) * HID;
  for (int i = tid; i < total; i += nthreads) {
    int pos  = i & 15;
    int lane = (i >> 4) & 31;
    int t    = i >> 9;
    int kt   = t % NKT;
    int n    = t / NKT;
    int k    = (kt << 5) + ((lane >> 4) << 3) + (pos & 7) + ((pos >> 3) << 4);
    int col  = (n << 4) + (lane & 15);
    dst[i] = (_Float16)W[k * HID + col];
  }
}

__device__ __forceinline__ v16h load_b_frag(const _Float16* __restrict__ Wf,
                                            int t, int lane) {
  return *(const v16h*)(Wf + (((t << 5) + lane) << 4));  // contiguous 32B
}

// --------------------------------------------------------------------------
// msg = nodes @ W_node  (no bias/activation), fp32 out. One wave -> 16 rows.
// --------------------------------------------------------------------------
__global__ __launch_bounds__(256) void node_proj_kernel(
    const float* __restrict__ nodes, const float* __restrict__ Wn,
    float* __restrict__ msg, int N) {
  extern __shared__ char smem_raw[];
  _Float16* sWf = (_Float16*)smem_raw;  // 128x128 f16 fragment-major = 32KB
  fill_b_frags<4>(sWf, Wn, threadIdx.x, blockDim.x);
  __syncthreads();

  const int wave = threadIdx.x >> 5;
  const int lane = threadIdx.x & 31;
  const int hi   = lane >> 4;
  const int c16  = lane & 15;

  const int r0 = (blockIdx.x * NWAVES + wave) * 16;
  if (r0 >= N) return;  // wave-uniform: EXEC all-ones for WMMA

  const float* arow = nodes + imin(r0 + c16, N - 1) * NODE_CH;
  v16h a[4];
#pragma unroll
  for (int kt = 0; kt < 4; ++kt) a[kt] = load_a_f32v(arow, kt * 32, hi);

#pragma unroll
  for (int n = 0; n < 8; ++n) {
    v16h bf[4];
#pragma unroll
    for (int kt = 0; kt < 4; ++kt) bf[kt] = load_b_frag(sWf, n * 4 + kt, lane);
    v8f acc = {};
#pragma unroll
    for (int kt = 0; kt < 4; ++kt) acc = wmma_f16(a[kt], bf[kt], acc);
    // D layout: lane (hi,c16), vgpr v -> element [v + 8*hi][n*16+c16]
#pragma unroll
    for (int v = 0; v < 8; ++v)
      msg[imin(r0 + v + 8 * hi, N - 1) * HID + n * 16 + c16] = acc[v];
  }
}

// --------------------------------------------------------------------------
// Fused: e = LReLU(LReLU(edges@W1+b1)@W2+b2); out[seg[e]] += msg[idx[e]] * e
// One wave -> 16 edges. Layer-1 result staged f16 row-major in per-wave LDS
// (scattered b16 stores, vectorized reads), re-fragmented as layer-2 A.
// Gather + f32 atomics straight from D-layout registers (64B segments, L2).
// --------------------------------------------------------------------------
__global__ __launch_bounds__(256) void edge_mlp_scatter_kernel(
    const float* __restrict__ edges,
    const float* __restrict__ W1, const float* __restrict__ b1,
    const float* __restrict__ W2, const float* __restrict__ b2,
    const float* __restrict__ msg,
    const int* __restrict__ srcIdx, const int* __restrict__ dstIdx,
    float* __restrict__ out, int E) {
  extern __shared__ char smem_raw[];
  _Float16* sW1f = (_Float16*)smem_raw;       // 64x128  f16 frag-major = 16KB
  _Float16* sW2f = sW1f + EDGE_CH * HID;      // 128x128 f16 frag-major = 32KB
  _Float16* sHid = sW2f + HID * HID;          // EWAVES x 16x128 f16 = 32KB

  fill_b_frags<2>(sW1f, W1, threadIdx.x, blockDim.x);
  fill_b_frags<4>(sW2f, W2, threadIdx.x, blockDim.x);
  __syncthreads();

  const int wave = threadIdx.x >> 5;
  const int lane = threadIdx.x & 31;
  const int hi   = lane >> 4;
  const int c16  = lane & 15;

  const int e0 = (blockIdx.x * EWAVES + wave) * 16;
  if (e0 >= E) return;  // wave-uniform

  // per-lane biases (8 n-tiles each), coalesced 64B global segments
  float bias1r[8], bias2r[8];
#pragma unroll
  for (int n = 0; n < 8; ++n) {
    bias1r[n] = b1[n * 16 + c16];
    bias2r[n] = b2[n * 16 + c16];
  }

  _Float16* hidRow = sHid + wave * 16 * HID;

  // ---- layer 1: [16x64] @ [64x128] ----
  const float* arow = edges + imin(e0 + c16, E - 1) * EDGE_CH;
  v16h a0 = load_a_f32v(arow, 0, hi);
  v16h a1 = load_a_f32v(arow, 32, hi);

#pragma unroll
  for (int n = 0; n < 8; ++n) {
    v16h bfA = load_b_frag(sW1f, n * 2 + 0, lane);
    v16h bfB = load_b_frag(sW1f, n * 2 + 1, lane);
    v8f acc = {};
    acc = wmma_f16(a0, bfA, acc);
    acc = wmma_f16(a1, bfB, acc);
    _Float16* hp = hidRow + (8 * hi) * HID + n * 16 + c16;
#pragma unroll
    for (int v = 0; v < 8; ++v)
      hp[v * HID] = (_Float16)lrelu(acc[v] + bias1r[n]);
  }
  // cross-lane LDS reuse below: ensure this wave's DS stores completed
  asm volatile("s_wait_dscnt 0x0" ::: "memory");

  // layer-2 A fragments, hoisted (read once, used for all 8 n-tiles)
  const _Float16* aRow2 = hidRow + c16 * HID;
  v16h a2[4];
#pragma unroll
  for (int kt = 0; kt < 4; ++kt) a2[kt] = load_a_f16v(aRow2, kt * 32, hi);

  // src/dst element offsets for the 8 rows this lane owns in D layout
  int sOff[8], dOff[8];  // <= 50000*128, fits int32
#pragma unroll
  for (int v = 0; v < 8; ++v) {
    int e = imin(e0 + v + 8 * hi, E - 1);
    sOff[v] = srcIdx[e] * HID;
    dOff[v] = dstIdx[e] * HID;
  }

  // ---- layer 2: [16x128] @ [128x128], then gate + scatter-add ----
#pragma unroll
  for (int n = 0; n < 8; ++n) {
    v16h bf[4];
#pragma unroll
    for (int kt = 0; kt < 4; ++kt) bf[kt] = load_b_frag(sW2f, n * 4 + kt, lane);
    v8f acc = {};
#pragma unroll
    for (int kt = 0; kt < 4; ++kt) acc = wmma_f16(a2[kt], bf[kt], acc);
    const int col = n * 16 + c16;
#pragma unroll
    for (int v = 0; v < 8; ++v) {
      float x = lrelu(acc[v] + bias2r[n]);
      atomicAdd(out + dOff[v] + col, msg[sOff[v] + col] * x);
    }
  }
}

// --------------------------------------------------------------------------
extern "C" void kernel_launch(void* const* d_in, const int* in_sizes, int n_in,
                              void* d_out, int out_size, void* d_ws,
                              size_t ws_size, hipStream_t stream) {
  const float* nodes = (const float*)d_in[0];
  const float* edges = (const float*)d_in[1];
  const int*   seg   = (const int*)d_in[2];
  const int*   idx   = (const int*)d_in[3];
  const float* Wn    = (const float*)d_in[4];
  const float* W1    = (const float*)d_in[5];
  const float* B1    = (const float*)d_in[6];
  const float* W2    = (const float*)d_in[7];
  const float* B2    = (const float*)d_in[8];

  const int N = in_sizes[0] / NODE_CH;  // 50000
  const int E = in_sizes[1] / EDGE_CH;  // 600000

  float* msg = (float*)d_ws;  // [N, HID] fp32 = 25.6MB

  hipMemsetAsync(d_out, 0, (size_t)out_size * sizeof(float), stream);

  {  // node projection
    int tiles  = (N + 15) / 16;
    int blocks = (tiles + NWAVES - 1) / NWAVES;
    size_t shmem = (size_t)NODE_CH * HID * sizeof(_Float16);
    node_proj_kernel<<<blocks, 256, shmem, stream>>>(nodes, Wn, msg, N);
  }
  {  // fused edge MLP + gather + gate + scatter-add
    int tiles  = (E + 15) / 16;
    int blocks = (tiles + EWAVES - 1) / EWAVES;
    size_t shmem =
        (size_t)(EDGE_CH * HID + HID * HID + EWAVES * 16 * HID) * sizeof(_Float16);
    edge_mlp_scatter_kernel<<<blocks, 256, shmem, stream>>>(
        edges, W1, B1, W2, B2, msg, idx, seg, (float*)d_out, E);
  }
}
